// SAGEConvPure_2817498546896
// MI455X (gfx1250) — matrix operand order repaired and checked by reading, not compiled, not run on hardware
//
#include <hip/hip_runtime.h>
#include <hip/hip_bf16.h>

typedef __attribute__((ext_vector_type(16))) _Float16 v16h;
typedef __attribute__((ext_vector_type(8)))  float    v8f;

#define D_IN   128
#define KDIM   256   // 2 * D_IN
#define D_OUT  128

// ---------------------------------------------------------------------------
// Zero the workspace (agg + count are contiguous)
// ---------------------------------------------------------------------------
__global__ void sage_zero_kernel(float* __restrict__ p, long long n) {
    long long i = (long long)blockIdx.x * blockDim.x + threadIdx.x;
    long long stride = (long long)gridDim.x * blockDim.x;
    for (; i < n; i += stride) p[i] = 0.0f;
}

// ---------------------------------------------------------------------------
// Scatter-add: one wave32 per edge. Row of x is 128 f32 = 32 lanes x float4.
// agg and x both fit in the 192MB L2, so atomics stay on-chip.
// ---------------------------------------------------------------------------
__global__ void sage_scatter_kernel(const float* __restrict__ x,
                                    const int* __restrict__ src,
                                    const int* __restrict__ dst,
                                    float* __restrict__ agg,
                                    float* __restrict__ cnt,
                                    int E) {
    const int lane   = threadIdx.x & 31;
    const int wave   = blockIdx.x * (blockDim.x >> 5) + (threadIdx.x >> 5);
    const int nwaves = gridDim.x * (blockDim.x >> 5);
    for (int e = wave; e < E; e += nwaves) {
        const int s = src[e];
        const int d = dst[e];
        const float4 v = ((const float4*)(x + (size_t)s * D_IN))[lane];
        float* ap = agg + (size_t)d * D_IN + lane * 4;
        unsafeAtomicAdd(ap + 0, v.x);
        unsafeAtomicAdd(ap + 1, v.y);
        unsafeAtomicAdd(ap + 2, v.z);
        unsafeAtomicAdd(ap + 3, v.w);
        if (lane == 0) unsafeAtomicAdd(cnt + d, 1.0f);
    }
}

// ---------------------------------------------------------------------------
// out = [x | agg/max(cnt,1)] @ W^T + b   via v_wmma_f32_16x16x32_f16.
// One wave per 16-row tile; 8 accumulator tiles cover all 128 output cols.
// W staged in LDS as f16 (64KB of the 320KB/WGP pool).
// ---------------------------------------------------------------------------
__global__ void sage_gemm_wmma_kernel(const float* __restrict__ x,
                                      const float* __restrict__ agg,
                                      const float* __restrict__ cnt,
                                      const float* __restrict__ W,
                                      const float* __restrict__ bias,
                                      float* __restrict__ out,
                                      int N, int nTiles) {
    __shared__ _Float16 Wlds[D_OUT * KDIM];   // 32768 halves = 64 KB

    const int tid = threadIdx.x;

    // Stage W (f32 row-major (128,256)) into LDS as f16, coalesced float4 loads.
    const float4* W4 = (const float4*)W;
    for (int i = tid; i < (D_OUT * KDIM) / 4; i += blockDim.x) {
        float4 w = W4[i];
        Wlds[4 * i + 0] = (_Float16)w.x;
        Wlds[4 * i + 1] = (_Float16)w.y;
        Wlds[4 * i + 2] = (_Float16)w.z;
        Wlds[4 * i + 3] = (_Float16)w.w;
    }
    __syncthreads();

    const int lane    = tid & 31;
    const int wave    = tid >> 5;
    const int rowTile = blockIdx.x * (blockDim.x >> 5) + wave;
    if (rowTile >= nTiles) return;           // uniform per wave: EXEC stays all-1s
    const int rowBase = rowTile * 16;

    const int half16 = lane >> 4;            // 0: lanes 0-15, 1: lanes 16-31
    const int lm     = lane & 15;

    int row = rowBase + lm;
    if (row >= N) row = N - 1;               // clamp loads; stores guarded below
    const float* rowx = x   + (size_t)row * D_IN;
    const float* rowa = agg + (size_t)row * D_IN;
    const float  c    = cnt[row];
    const float  invc = 1.0f / (c > 1.0f ? c : 1.0f);

    v8f acc[8];
#pragma unroll
    for (int nt = 0; nt < 8; ++nt) acc[nt] = (v8f){0.f,0.f,0.f,0.f,0.f,0.f,0.f,0.f};

    float biasr[8];
#pragma unroll
    for (int nt = 0; nt < 8; ++nt) biasr[nt] = bias[nt * 16 + lm];

    const int kbaseA = half16 * 16;          // A frag: 16 contiguous K values per lane
    const int kbaseB = half16 * 16;          // B frag: 16 contiguous K values per lane

#pragma unroll
    for (int kk = 0; kk < KDIM; kk += 32) {
        // Columns [kk, kk+32) come entirely from x (kk<128) or from agg/cnt.
        const float* rsrc;
        float scale;
        if (kk < D_IN) { rsrc = rowx + kk;          scale = 1.0f; }
        else           { rsrc = rowa + (kk - D_IN); scale = invc; }

        const float4 p0 = *(const float4*)(rsrc + kbaseA);
        const float4 p1 = *(const float4*)(rsrc + kbaseA + 4);
        const float4 p2 = *(const float4*)(rsrc + kbaseA + 8);
        const float4 p3 = *(const float4*)(rsrc + kbaseA + 12);
        const float fa[16] = { p0.x, p0.y, p0.z, p0.w,  p1.x, p1.y, p1.z, p1.w,
                               p2.x, p2.y, p2.z, p2.w,  p3.x, p3.y, p3.z, p3.w };
        v16h a;
#pragma unroll
        for (int i = 0; i < 16; ++i) a[i] = (_Float16)(fa[i] * scale);

#pragma unroll
        for (int nt = 0; nt < 8; ++nt) {
            // B[k][n] = W[n][k]; lane holds column n = nt*16+lm, K range kk+kbaseB..+15
            const v16h bf = *(const v16h*)(Wlds + (size_t)(nt * 16 + lm) * KDIM + kk + kbaseB);
            acc[nt] = __builtin_amdgcn_wmma_f32_16x16x32_f16(
                false, a, false, bf, (short)0, acc[nt], false, false);
        }
    }

    // C/D layout: VGPR r -> M = r + half16*8, N = lane&15
    const int rowOff = half16 * 8;
#pragma unroll
    for (int r = 0; r < 8; ++r) {
        const int orow = rowBase + rowOff + r;
        if (orow < N) {
            float* op = out + (size_t)orow * D_OUT + lm;
#pragma unroll
            for (int nt = 0; nt < 8; ++nt) op[nt * 16] = acc[nt][r] + biasr[nt];
        }
    }
}

// ---------------------------------------------------------------------------
extern "C" void kernel_launch(void* const* d_in, const int* in_sizes, int n_in,
                              void* d_out, int out_size, void* d_ws, size_t ws_size,
                              hipStream_t stream) {
    const float* x  = (const float*)d_in[0];
    const int*   ei = (const int*)d_in[1];
    const float* W  = (const float*)d_in[2];
    const float* b  = (const float*)d_in[3];

    const int N = in_sizes[0] / D_IN;        // 100000
    const int E = in_sizes[1] / 2;           // 3200000
    const int* src = ei;
    const int* dst = ei + E;

    float* agg = (float*)d_ws;               // N*128 f32
    float* cnt = agg + (size_t)N * D_IN;     // N f32

    // 1) zero agg + count (contiguous N*129 floats)
    const long long zn = (long long)N * (D_IN + 1);
    sage_zero_kernel<<<2048, 256, 0, stream>>>(agg, zn);

    // 2) scatter-mean numerator + degree counts (one wave per edge, grid-stride)
    sage_scatter_kernel<<<12288, 256, 0, stream>>>(x, src, dst, agg, cnt, E);

    // 3) fused concat + GEMM + bias via WMMA
    const int nTiles  = (N + 15) / 16;       // 6250
    const int wavesPB = 8;                   // 256 threads / 32
    const int blocks  = (nTiles + wavesPB - 1) / wavesPB;
    sage_gemm_wmma_kernel<<<blocks, 256, 0, stream>>>(x, agg, cnt, W, b,
                                                      (float*)d_out, N, nTiles);
}
